// SelfAttention_52518860096315
// MI455X (gfx1250) — matrix operand order repaired and checked by reading, not compiled, not run on hardware
//
#include <hip/hip_runtime.h>
#include <hip/hip_bf16.h>
#include <math.h>

typedef __attribute__((ext_vector_type(16))) __bf16 v16bf;
typedef __attribute__((ext_vector_type(8)))  float  v8f;
typedef __attribute__((ext_vector_type(4)))  int    v4i;

#define DDEPTH 128
#define SEQ    1024
#define NBH    80              // B*H = 10*8
#define NROWS  (NBH * SEQ)     // 81920

#if __has_builtin(__builtin_amdgcn_global_load_async_to_lds_b128)
#define USE_ASYNC_LDS 1
#else
#define USE_ASYNC_LDS 0
#endif
#if __has_builtin(__builtin_amdgcn_s_wait_asynccnt)
#define HAVE_WAIT_ASYNC 1
#else
#define HAVE_WAIT_ASYNC 0
#endif

union BfFrag { v16bf v; uint4 u[2]; };

// 8 contiguous f32 -> bf16 into fragment elements [base .. base+7]
__device__ __forceinline__ void cvt8(const float* p, v16bf& v, int base) {
    float4 a = *(const float4*)p;
    float4 b = *(const float4*)(p + 4);
    v[base + 0] = (__bf16)a.x; v[base + 1] = (__bf16)a.y;
    v[base + 2] = (__bf16)a.z; v[base + 3] = (__bf16)a.w;
    v[base + 4] = (__bf16)b.x; v[base + 5] = (__bf16)b.y;
    v[base + 6] = (__bf16)b.z; v[base + 7] = (__bf16)b.w;
}
// 16 contiguous f32 -> full bf16 fragment (contiguous 16-K order; unused, B operands here use the split K order)
__device__ __forceinline__ v16bf cvt16(const float* p) {
    v16bf v; cvt8(p, v, 0); cvt8(p + 8, v, 8); return v;
}
// 16 contiguous bf16 (two b128 loads; works for global or LDS pointers)
__device__ __forceinline__ v16bf ld16bf(const __bf16* p) {
    BfFrag f; f.u[0] = *(const uint4*)p; f.u[1] = *(const uint4*)(p + 8); return f.v;
}
// two separate 8-element bf16 runs (A-operand K split {0..7, 16..23})
__device__ __forceinline__ v16bf ld8x2bf(const __bf16* p0, const __bf16* p1) {
    BfFrag f; f.u[0] = *(const uint4*)p0; f.u[1] = *(const uint4*)p1; return f.v;
}

// 16-byte global -> LDS copy: async (ASYNCcnt-tracked) on CDNA5 toolchains
__device__ __forceinline__ void cp16(const __bf16* g, __bf16* l) {
#if USE_ASYNC_LDS
    __builtin_amdgcn_global_load_async_to_lds_b128(
        (__attribute__((address_space(1))) v4i*)(g),
        (__attribute__((address_space(3))) v4i*)(l), 0, 0);
#else
    *(uint4*)l = *(const uint4*)g;
#endif
}
__device__ __forceinline__ void async_wait0() {
#if USE_ASYNC_LDS
#if HAVE_WAIT_ASYNC
    __builtin_amdgcn_s_wait_asynccnt(0);
#else
    asm volatile("s_wait_asynccnt 0" ::: "memory");
#endif
#endif
}

// ---------------------------------------------------------------------------
// Projection GEMM: Y[m][n] = sum_k A[m][k] * W[n][k] + bias[n]
// MODE 0: A = f32, store bf16 row-major (q, k)
// MODE 1: A = f32, store bf16 transposed per (b,h): vT[bh][d][s2]   (v)
// MODE 2: A = bf16 (attn out), store f32 with (b,h,s)->(b,s,h) remap (wo)
// ---------------------------------------------------------------------------
template <int MODE>
__global__ __launch_bounds__(256)
void proj_kernel(const float* __restrict__ Xf, const __bf16* __restrict__ Xb,
                 const float* __restrict__ W, const float* __restrict__ bias,
                 __bf16* __restrict__ outb, float* __restrict__ outf)
{
    const int lane = threadIdx.x & 31;
    const int wave = threadIdx.x >> 5;       // n-tile 0..7
    const int l16  = lane & 15;
    const int hl   = lane >> 4;              // lane half
    const int row0 = blockIdx.x * 16;
    const int ncol = wave * 16 + l16;
    const int arow = row0 + l16;

    v8f c = {};
#pragma unroll
    for (int kc = 0; kc < 4; ++kc) {
        const int ka = kc * 32 + hl * 8;     // A-layout K base
        const int kb = kc * 32 + hl * 8;     // B-layout K base
        v16bf a, b;
        if (MODE == 2) {
            const __bf16* ap = Xb + (size_t)arow * DDEPTH;
            a = ld8x2bf(ap + ka, ap + ka + 16);
        } else {
            const float* ap = Xf + (size_t)arow * DDEPTH;
            cvt8(ap + ka, a, 0); cvt8(ap + ka + 16, a, 8);
        }
        cvt8(W + (size_t)ncol * DDEPTH + kb, b, 0); cvt8(W + (size_t)ncol * DDEPTH + kb + 16, b, 8);
        c = __builtin_amdgcn_wmma_f32_16x16x32_bf16(false, a, false, b,
                                                    (short)0, c, false, false);
    }
    const float bv = bias[ncol];
#pragma unroll
    for (int r = 0; r < 8; ++r) {
        const int m = row0 + hl * 8 + r;     // C/D layout: M = 8*half + vgpr
        const float o = c[r] + bv;
        if (MODE == 0) {
            outb[(size_t)m * DDEPTH + ncol] = (__bf16)o;
        } else if (MODE == 1) {
            const int bh = m >> 10, s = m & 1023;
            outb[((size_t)bh * DDEPTH + ncol) * SEQ + s] = (__bf16)o;
        } else {
            const int bh = m >> 10, s = m & 1023;
            const int bb = bh >> 3, h = bh & 7;
            outf[(((size_t)bb * SEQ + s) * 8 + h) * DDEPTH + ncol] = o;
        }
    }
}

// ---------------------------------------------------------------------------
// Flash attention v2: 256-thread workgroup = 8 waves = 128 queries of one
// (b,h). The workgroup cooperatively stages a 64-key K tile (16KB) and V tile
// (16KB) in LDS via async-to-LDS, shared by all 8 waves (8x less K/V traffic
// than per-wave streaming). Each wave: 16 score WMMAs + online softmax
// (cross-lane tree once per 64 keys) + P transpose via per-wave LDS region +
// 16 PV WMMAs into a 16x128 f32 register tile.
// ---------------------------------------------------------------------------
__global__ __launch_bounds__(256)
void flash_attn_kernel(const __bf16* __restrict__ qg, const __bf16* __restrict__ kg,
                       const __bf16* __restrict__ vT, __bf16* __restrict__ ao)
{
    __shared__ __bf16 Kt[64 * 128];          // [key][d]     16 KB
    __shared__ __bf16 Vt[128 * 64];          // [d][key]     16 KB
    __shared__ __bf16 Pt[8 * 16 * 64];       // per-wave P   16 KB

    const int tid  = threadIdx.x;
    const int lane = tid & 31;
    const int wave = tid >> 5;               // 0..7
    const int l16  = lane & 15;
    const int hl   = lane >> 4;
    const int bh    = blockIdx.y;            // 0..79
    const int qrow0 = blockIdx.x * 128 + wave * 16;

    // Q fragments (A operand) for 4 depth chunks, pinned in registers
    v16bf qf[4];
    const __bf16* qrow = qg + ((size_t)bh * SEQ + qrow0 + l16) * DDEPTH;
#pragma unroll
    for (int kc = 0; kc < 4; ++kc) {
        const int ka = kc * 32 + hl * 8;
        qf[kc] = ld8x2bf(qrow + ka, qrow + ka + 16);
    }

    float mr[8], lr[8];
#pragma unroll
    for (int r = 0; r < 8; ++r) { mr[r] = -INFINITY; lr[r] = 0.f; }
    v8f acc[8] = {};

    const __bf16* kbase = kg + (size_t)bh * SEQ * DDEPTH;   // [s2][d]
    const __bf16* vbase = vT + (size_t)bh * DDEPTH * SEQ;   // [d][s2]
    __bf16* Pw = Pt + wave * (16 * 64);

    for (int ch = 0; ch < SEQ; ch += 64) {
        // ---- cooperative async stage of K (contiguous) and V (row slices) ----
        {
            const __bf16* gk = kbase + (size_t)ch * DDEPTH + tid * 32;  // 64B/thread
            __bf16* lk = Kt + tid * 32;
            const int vr = tid >> 1, vh = tid & 1;
            const __bf16* gv = vbase + (size_t)vr * SEQ + ch + vh * 32; // 64B/thread
            __bf16* lv = Vt + vr * 64 + vh * 32;
#pragma unroll
            for (int j = 0; j < 4; ++j) cp16(gk + j * 8, lk + j * 8);
#pragma unroll
            for (int j = 0; j < 4; ++j) cp16(gv + j * 8, lv + j * 8);
        }
        async_wait0();
        __syncthreads();

        // ---- scores: 16 q x 64 keys, B operands from LDS ----
        v8f c0 = {}, c1 = {}, c2 = {}, c3 = {};
#pragma unroll
        for (int kc = 0; kc < 4; ++kc) {
            const int kd = kc * 32 + hl * 8;
            v16bf b0 = ld8x2bf(Kt + (l16 +  0) * DDEPTH + kd, Kt + (l16 +  0) * DDEPTH + kd + 16);
            v16bf b1 = ld8x2bf(Kt + (l16 + 16) * DDEPTH + kd, Kt + (l16 + 16) * DDEPTH + kd + 16);
            v16bf b2 = ld8x2bf(Kt + (l16 + 32) * DDEPTH + kd, Kt + (l16 + 32) * DDEPTH + kd + 16);
            v16bf b3 = ld8x2bf(Kt + (l16 + 48) * DDEPTH + kd, Kt + (l16 + 48) * DDEPTH + kd + 16);
            c0 = __builtin_amdgcn_wmma_f32_16x16x32_bf16(false, qf[kc], false, b0, (short)0, c0, false, false);
            c1 = __builtin_amdgcn_wmma_f32_16x16x32_bf16(false, qf[kc], false, b1, (short)0, c1, false, false);
            c2 = __builtin_amdgcn_wmma_f32_16x16x32_bf16(false, qf[kc], false, b2, (short)0, c2, false, false);
            c3 = __builtin_amdgcn_wmma_f32_16x16x32_bf16(false, qf[kc], false, b3, (short)0, c3, false, false);
        }
        // ---- online softmax: one cross-lane tree per row per 64 keys ----
        float corr[8];
#pragma unroll
        for (int r = 0; r < 8; ++r) {
            float s0 = c0[r], s1 = c1[r], s2 = c2[r], s3 = c3[r];
            float tm = fmaxf(fmaxf(s0, s1), fmaxf(s2, s3));
#pragma unroll
            for (int w = 1; w < 16; w <<= 1) tm = fmaxf(tm, __shfl_xor(tm, w, 16));
            const float mnew = fmaxf(mr[r], tm);
            corr[r] = __expf(mr[r] - mnew);
            const float p0 = __expf(s0 - mnew);
            const float p1 = __expf(s1 - mnew);
            const float p2 = __expf(s2 - mnew);
            const float p3 = __expf(s3 - mnew);
            float rs = (p0 + p1) + (p2 + p3);
#pragma unroll
            for (int w = 1; w < 16; w <<= 1) rs += __shfl_xor(rs, w, 16);
            lr[r] = lr[r] * corr[r] + rs;
            mr[r] = mnew;
            __bf16* pp = Pw + (hl * 8 + r) * 64 + l16;   // D-layout -> [row][key]
            pp[0]  = (__bf16)p0;
            pp[16] = (__bf16)p1;
            pp[32] = (__bf16)p2;
            pp[48] = (__bf16)p3;
        }
#pragma unroll
        for (int dn = 0; dn < 8; ++dn)
#pragma unroll
            for (int r = 0; r < 8; ++r) acc[dn][r] *= corr[r];

        // ---- P: reload in A-layout (per-wave region; wave-local ordering) ----
        const __bf16* pr = Pw + l16 * 64;
        v16bf pf0 = ld8x2bf(pr + hl * 8,      pr + hl * 8 + 16);
        v16bf pf1 = ld8x2bf(pr + 32 + hl * 8, pr + 32 + hl * 8 + 16);

        // ---- acc += P @ V, B operands from LDS V tile ----
#pragma unroll
        for (int dn = 0; dn < 8; ++dn) {
            const __bf16* vp = Vt + (dn * 16 + l16) * 64 + hl * 8;
            v16bf vb0 = ld8x2bf(vp, vp + 16);
            v16bf vb1 = ld8x2bf(vp + 32, vp + 48);
            acc[dn] = __builtin_amdgcn_wmma_f32_16x16x32_bf16(false, pf0, false, vb0, (short)0, acc[dn], false, false);
            acc[dn] = __builtin_amdgcn_wmma_f32_16x16x32_bf16(false, pf1, false, vb1, (short)0, acc[dn], false, false);
        }
        __syncthreads();   // protect Kt/Vt before next chunk's stage
    }

    // ---- normalize and store bf16 attn-out row-major ----
#pragma unroll
    for (int dn = 0; dn < 8; ++dn) {
#pragma unroll
        for (int r = 0; r < 8; ++r) {
            const int mrow = hl * 8 + r;
            const float o = acc[dn][r] / lr[r];
            ao[((size_t)bh * SEQ + qrow0 + mrow) * DDEPTH + dn * 16 + l16] = (__bf16)o;
        }
    }
}

extern "C" void kernel_launch(void* const* d_in, const int* in_sizes, int n_in,
                              void* d_out, int out_size, void* d_ws, size_t ws_size,
                              hipStream_t stream) {
    const float* x1   = (const float*)d_in[0];
    const float* x2   = (const float*)d_in[1];
    const float* wq_w = (const float*)d_in[2];
    const float* wq_b = (const float*)d_in[3];
    const float* wk_w = (const float*)d_in[4];
    const float* wk_b = (const float*)d_in[5];
    const float* wv_w = (const float*)d_in[6];
    const float* wv_b = (const float*)d_in[7];
    const float* wo_w = (const float*)d_in[8];
    const float* wo_b = (const float*)d_in[9];
    float* out = (float*)d_out;

    const size_t E = (size_t)NROWS * DDEPTH;          // 10,485,760 elements
    __bf16* qb = (__bf16*)d_ws;                       // bf16 q   (20 MB)
    __bf16* kb = qb + E;                              // bf16 k   (20 MB)
    __bf16* vT = kb + E;                              // bf16 v^T (20 MB)
    __bf16* ao = vT + E;                              // bf16 attn out (20 MB)

    const dim3 pgrid(NROWS / 16), pblk(256);
    proj_kernel<0><<<pgrid, pblk, 0, stream>>>(x1, nullptr, wq_w, wq_b, qb, nullptr);
    proj_kernel<0><<<pgrid, pblk, 0, stream>>>(x2, nullptr, wk_w, wk_b, kb, nullptr);
    proj_kernel<1><<<pgrid, pblk, 0, stream>>>(x2, nullptr, wv_w, wv_b, vT, nullptr);

    flash_attn_kernel<<<dim3(SEQ / 128, NBH), dim3(256), 0, stream>>>(qb, kb, vT, ao);

    proj_kernel<2><<<pgrid, pblk, 0, stream>>>(nullptr, ao, wo_w, wo_b, nullptr, out);
}
